// DANetHead_1168231105168
// MI455X (gfx1250) — matrix operand order repaired and checked by reading, not compiled, not run on hardware
//
#include <hip/hip_runtime.h>
#include <hip/hip_bf16.h>
#include <math.h>

// ---------------------------------------------------------------------------
// DANet head for MI455X (gfx1250): WMMA f16->f32 GEMM pipeline.
// Round 2: 2x2 register blocking (32x32 per wave) on the two dominant GEMMs.
// ---------------------------------------------------------------------------
constexpr int B_   = 4;
constexpr int C_   = 512;
constexpr int C8_  = 64;
constexpr int N_   = 4096;   // H*W
constexpr int OUT_ = 256;

typedef __attribute__((ext_vector_type(16))) _Float16 v16h;
typedef __attribute__((ext_vector_type(8)))  float    v8f;

#define WAVE ((int)(threadIdx.x >> 5))
#define LANE ((int)(threadIdx.x & 31))

// A-fragment 16x32 f16, row-major source (lda elements/row).
// Lane L: M = L%16 ; K = (L/16)*16 .. +15 contiguous.
__device__ __forceinline__ v16h load_a16(const _Float16* __restrict__ base,
                                         int lda, int row0, int k0, int lane) {
  const _Float16* p = base + (size_t)(row0 + (lane & 15)) * lda + k0 + ((lane >> 4) << 4);
  v16h a;
#pragma unroll
  for (int i = 0; i < 16; ++i) a[i] = p[i];
  return a;
}

// B-fragment 32x16 f16 loaded from B^T row-major (ldb elements/row).
// Lane L: N = L%16 ; K = (L/16)*16 .. +15 contiguous.
__device__ __forceinline__ v16h load_bT16(const _Float16* __restrict__ bT,
                                          int ldb, int col0, int k0, int lane) {
  const _Float16* p = bT + (size_t)(col0 + (lane & 15)) * ldb + k0 + ((lane >> 4) << 4);
  v16h b;
#pragma unroll
  for (int i = 0; i < 16; ++i) b[i] = p[i];
  return b;
}

// B-fragment from a per-lane row pointer (conv halo) with zero fill.
__device__ __forceinline__ v16h load_row16(const _Float16* __restrict__ rowbase,
                                           int k0, int lane, bool valid) {
  v16h b;
  if (valid) {
    const _Float16* p = rowbase + k0 + ((lane >> 4) << 4);
#pragma unroll
    for (int i = 0; i < 16; ++i) b[i] = p[i];
  } else {
#pragma unroll
    for (int i = 0; i < 16; ++i) b[i] = (_Float16)0.f;
  }
  return b;
}

__device__ __forceinline__ v8f wmma16(v16h a, v16h b, v8f c) {
  return __builtin_amdgcn_wmma_f32_16x16x32_f16(false, a, false, b, (short)0, c, false, false);
}

// ---------------------------------------------------------------------------
// Kernel 0: f32 -> f16 convert, both [C,N] and transposed [N,C] layouts.
__global__ void k_convert(const float* __restrict__ x,
                          _Float16* __restrict__ x16,
                          _Float16* __restrict__ xT16) {
  size_t idx = (size_t)blockIdx.x * blockDim.x + threadIdx.x;
  if (idx >= (size_t)B_ * C_ * N_) return;
  int n = (int)(idx % N_);
  int c = (int)((idx / N_) % C_);
  int b = (int)(idx / ((size_t)N_ * C_));
  _Float16 v = (_Float16)x[idx];
  x16[idx] = v;
  xT16[((size_t)b * N_ + n) * C_ + c] = v;
}

// Kernel 1: pack all weights to f16; conv weights reordered [9][OUT][C].
__global__ void k_pack(const float* __restrict__ wq, const float* __restrict__ wk,
                       const float* __restrict__ wv, const float* __restrict__ cw,
                       _Float16* __restrict__ wq16, _Float16* __restrict__ wk16,
                       _Float16* __restrict__ wv16, _Float16* __restrict__ wc16) {
  int tid = blockIdx.x * blockDim.x + threadIdx.x;
  const int NQ = C8_ * C_;        // 32768
  const int NV = C_ * C_;         // 262144
  const int NC = OUT_ * C_ * 9;   // 1179648
  if (tid < NQ) { wq16[tid] = (_Float16)wq[tid]; wk16[tid] = (_Float16)wk[tid]; }
  if (tid < NV) { wv16[tid] = (_Float16)wv[tid]; }
  if (tid < NC) {
    int o  = tid / (C_ * 9);
    int r  = tid % (C_ * 9);
    int c  = r / 9;
    int kk = r % 9;   // kh*3+kw
    wc16[((size_t)kk * OUT_ + o) * C_ + c] = (_Float16)cw[tid];
  }
}

// Kernel 2: q[b,n,i] and kT[b,n,i] = W(64x512) * x + bias, both stored [N,64] f16.
__global__ void k_qk(const _Float16* __restrict__ xT16,
                     const _Float16* __restrict__ wq16, const _Float16* __restrict__ wk16,
                     const float* __restrict__ bq, const float* __restrict__ bk,
                     _Float16* __restrict__ q16, _Float16* __restrict__ kT16) {
  int b    = blockIdx.z;
  int col0 = blockIdx.x * 16;                 // i (0..63)
  int row0 = blockIdx.y * 128 + WAVE * 16;    // n (0..4095)
  int lane = LANE;
  const _Float16* A = xT16 + (size_t)b * N_ * C_;
  v8f accq = {}; v8f acck = {};
  for (int k0 = 0; k0 < C_; k0 += 32) {
    v16h a  = load_a16(A, C_, row0, k0, lane);
    v16h fq = load_bT16(wq16, C_, col0, k0, lane);
    v16h fk = load_bT16(wk16, C_, col0, k0, lane);
    accq = wmma16(a, fq, accq);
    acck = wmma16(a, fk, acck);
  }
  int nc = col0 + (lane & 15);
  float biasq = bq[nc], biask = bk[nc];
#pragma unroll
  for (int i = 0; i < 8; ++i) {
    int m = row0 + i + ((lane >> 4) << 3);
    q16 [((size_t)b * N_ + m) * C8_ + nc] = (_Float16)(accq[i] + biasq);
    kT16[((size_t)b * N_ + m) * C8_ + nc] = (_Float16)(acck[i] + biask);
  }
}

// Kernel 3: v[b,c,n] = wv(512x512) * x + bv, stored [C,N] f16.
__global__ void k_v(const _Float16* __restrict__ xT16, const _Float16* __restrict__ wv16,
                    const float* __restrict__ bv, _Float16* __restrict__ vbuf) {
  int b    = blockIdx.z;
  int col0 = blockIdx.x * 16;                 // n
  int row0 = blockIdx.y * 128 + WAVE * 16;    // c
  int lane = LANE;
  const _Float16* BT = xT16 + (size_t)b * N_ * C_;
  v8f acc = {};
  for (int k0 = 0; k0 < C_; k0 += 32) {
    v16h a  = load_a16(wv16, C_, row0, k0, lane);
    v16h bb = load_bT16(BT, C_, col0, k0, lane);
    acc = wmma16(a, bb, acc);
  }
  int n = col0 + (lane & 15);
#pragma unroll
  for (int i = 0; i < 8; ++i) {
    int c = row0 + i + ((lane >> 4) << 3);
    vbuf[((size_t)b * C_ + c) * N_ + n] = (_Float16)(acc[i] + bv[c]);
  }
}

// Kernel 4: energy[b,m,n] = q[m,:]·kT[n,:]  (K = 64), stored f16 (softmaxed in place).
__global__ void k_energy(const _Float16* __restrict__ q16, const _Float16* __restrict__ kT16,
                         _Float16* __restrict__ attn) {
  int b    = blockIdx.z;
  int col0 = blockIdx.x * 16;                 // n
  int row0 = blockIdx.y * 128 + WAVE * 16;    // m
  int lane = LANE;
  const _Float16* A  = q16  + (size_t)b * N_ * C8_;
  const _Float16* BT = kT16 + (size_t)b * N_ * C8_;
  v8f acc = {};
  for (int k0 = 0; k0 < C8_; k0 += 32) {
    v16h a  = load_a16(A, C8_, row0, k0, lane);
    v16h bb = load_bT16(BT, C8_, col0, k0, lane);
    acc = wmma16(a, bb, acc);
  }
  int n = col0 + (lane & 15);
#pragma unroll
  for (int i = 0; i < 8; ++i) {
    int m = row0 + i + ((lane >> 4) << 3);
    attn[((size_t)b * N_ + m) * N_ + n] = (_Float16)acc[i];
  }
}

// Kernel 5: in-place row softmax over 4096 f16 logits. One block per row.
__global__ void k_softmax(_Float16* __restrict__ attn) {
  __shared__ float red[256];
  size_t row = blockIdx.x;                    // b*N + m
  _Float16* p = attn + row * N_;
  int tid = threadIdx.x;
  float v[16];
#pragma unroll
  for (int i = 0; i < 16; ++i) v[i] = (float)p[tid * 16 + i];
  float mx = -1e30f;
#pragma unroll
  for (int i = 0; i < 16; ++i) mx = fmaxf(mx, v[i]);
  red[tid] = mx; __syncthreads();
  for (int s = 128; s > 0; s >>= 1) {
    if (tid < s) red[tid] = fmaxf(red[tid], red[tid + s]);
    __syncthreads();
  }
  mx = red[0]; __syncthreads();
  float sum = 0.f;
#pragma unroll
  for (int i = 0; i < 16; ++i) { v[i] = __expf(v[i] - mx); sum += v[i]; }
  red[tid] = sum; __syncthreads();
  for (int s = 128; s > 0; s >>= 1) {
    if (tid < s) red[tid] += red[tid + s];
    __syncthreads();
  }
  float inv = 1.f / red[0];
#pragma unroll
  for (int i = 0; i < 16; ++i) p[tid * 16 + i] = (_Float16)(v[i] * inv);
}

// Kernel 6: pa[b,c,m] = sum_n v[c,n]*attn[m,n] + x[b,c,m]   (dominant GEMM, K = 4096).
// 2x2 register blocking: one wave owns a 32x32 output tile.
// A = v (C x N), B^T = attn (row m contiguous over n).
__global__ void k_pa(const _Float16* __restrict__ vbuf, const _Float16* __restrict__ attn,
                     const float* __restrict__ x, float* __restrict__ pa) {
  int b    = blockIdx.z;
  int col0 = blockIdx.x * 32;                 // m
  int row0 = blockIdx.y * 256 + WAVE * 32;    // c
  int lane = LANE;
  const _Float16* A  = vbuf + (size_t)b * C_ * N_;
  const _Float16* BT = attn + (size_t)b * (size_t)N_ * N_;
  const _Float16* bpre0 = BT + (size_t)(col0 + (lane & 15)) * N_;
  const _Float16* bpre1 = bpre0 + (size_t)16 * N_;
  v8f acc00 = {}, acc01 = {}, acc10 = {}, acc11 = {};
  for (int k0 = 0; k0 < N_; k0 += 32) {
    __builtin_prefetch(bpre0 + k0 + 512, 0, 1);   // global_prefetch_b8 ahead in K
    __builtin_prefetch(bpre1 + k0 + 512, 0, 1);
    v16h a0 = load_a16(A, N_, row0,      k0, lane);
    v16h a1 = load_a16(A, N_, row0 + 16, k0, lane);
    v16h b0 = load_bT16(BT, N_, col0,      k0, lane);
    v16h b1 = load_bT16(BT, N_, col0 + 16, k0, lane);
    acc00 = wmma16(a0, b0, acc00);
    acc01 = wmma16(a0, b1, acc01);
    acc10 = wmma16(a1, b0, acc10);
    acc11 = wmma16(a1, b1, acc11);
  }
  int m0 = col0 + (lane & 15);
#pragma unroll
  for (int i = 0; i < 8; ++i) {
    int c0 = row0 + i + ((lane >> 4) << 3);
    size_t i00 = ((size_t)b * C_ + c0) * N_ + m0;
    size_t i10 = ((size_t)b * C_ + c0 + 16) * N_ + m0;
    pa[i00]      = acc00[i] + x[i00];
    pa[i00 + 16] = acc01[i] + x[i00 + 16];
    pa[i10]      = acc10[i] + x[i10];
    pa[i10 + 16] = acc11[i] + x[i10 + 16];
  }
}

// Kernel 7: avg + max pool over N per (b,c). One block per row.
__global__ void k_pool(const float* __restrict__ pa, float* __restrict__ avg,
                       float* __restrict__ mxo) {
  __shared__ float rs[256];
  __shared__ float rm[256];
  size_t row = blockIdx.x;                    // b*C + c
  const float* p = pa + row * N_;
  int tid = threadIdx.x;
  float s = 0.f, m = -1e30f;
  for (int i = tid; i < N_; i += 256) { float v = p[i]; s += v; m = fmaxf(m, v); }
  rs[tid] = s; rm[tid] = m; __syncthreads();
  for (int st = 128; st > 0; st >>= 1) {
    if (tid < st) { rs[tid] += rs[tid + st]; rm[tid] = fmaxf(rm[tid], rm[tid + st]); }
    __syncthreads();
  }
  if (tid == 0) { avg[row] = rs[0] * (1.f / N_); mxo[row] = rm[0]; }
}

// Kernel 8: SE MLP: scale = sigmoid(w2·relu(w1·avg) + w2·relu(w1·max)). One block per batch.
__global__ void k_mlp(const float* __restrict__ avg, const float* __restrict__ mxp,
                      const float* __restrict__ w1, const float* __restrict__ w2,
                      float* __restrict__ scale) {
  __shared__ float ha[C8_];
  __shared__ float hm[C8_];
  int b = blockIdx.x, tid = threadIdx.x;
  if (tid < C8_) {
    float sa = 0.f, sm = 0.f;
    for (int c = 0; c < C_; ++c) {
      float w = w1[tid * C_ + c];
      sa += w * avg[b * C_ + c];
      sm += w * mxp[b * C_ + c];
    }
    ha[tid] = fmaxf(sa, 0.f); hm[tid] = fmaxf(sm, 0.f);
  }
  __syncthreads();
  float sa = 0.f, sm = 0.f;
  for (int i = 0; i < C8_; ++i) {
    float w = w2[tid * C8_ + i];
    sa += w * ha[i]; sm += w * hm[i];
  }
  scale[b * C_ + tid] = 1.f / (1.f + __expf(-(sa + sm)));
}

// Kernel 9: caT[b,n,c] = (f16)(scale[b,c] * pa[b,c,n])  (pixel-major conv input).
__global__ void k_caT(const float* __restrict__ pa, const float* __restrict__ scale,
                      _Float16* __restrict__ caT) {
  size_t idx = (size_t)blockIdx.x * blockDim.x + threadIdx.x;
  if (idx >= (size_t)B_ * C_ * N_) return;
  int n = (int)(idx % N_);
  int c = (int)((idx / N_) % C_);
  int b = (int)(idx / ((size_t)N_ * C_));
  caT[((size_t)b * N_ + n) * C_ + c] = (_Float16)(pa[idx] * scale[b * C_ + c]);
}

// Kernel 10: 3x3 conv as 9 shifted WMMA GEMMs (K = 9*512) + BN + ReLU.
// 2x2 register blocking: one wave owns 32 output channels x 32 pixels.
// A = wpack[kk] (OUT x C), B^T rows = caT pixel rows with halo zero-fill.
__global__ void k_conv(const _Float16* __restrict__ wc16, const _Float16* __restrict__ caT,
                       const float* __restrict__ gamma, const float* __restrict__ beta,
                       const float* __restrict__ mean, const float* __restrict__ var,
                       float* __restrict__ out) {
  int b    = blockIdx.z;
  int col0 = blockIdx.x * 32;                 // pixel n
  int row0 = WAVE * 32;                       // o (8 waves * 32 = 256 = OUT)
  int lane = LANE;
  const _Float16* BT0 = caT + (size_t)b * N_ * C_;
  int n0  = col0 + (lane & 15);
  int n1  = n0 + 16;
  int h0 = n0 >> 6, w0 = n0 & 63;
  int h1 = n1 >> 6, w1 = n1 & 63;
  v8f acc00 = {}, acc01 = {}, acc10 = {}, acc11 = {};
#pragma unroll
  for (int dh = -1; dh <= 1; ++dh) {
#pragma unroll
    for (int dw = -1; dw <= 1; ++dw) {
      int kk = (dh + 1) * 3 + (dw + 1);
      const _Float16* A = wc16 + (size_t)kk * OUT_ * C_;
      int ha = h0 + dh, wa = w0 + dw;
      int hb = h1 + dh, wb = w1 + dw;
      bool va = ((unsigned)ha < 64u) && ((unsigned)wa < 64u);
      bool vb = ((unsigned)hb < 64u) && ((unsigned)wb < 64u);
      int na = va ? (ha * 64 + wa) : 0;
      int nb = vb ? (hb * 64 + wb) : 0;
      const _Float16* rowa = BT0 + (size_t)na * C_;
      const _Float16* rowb = BT0 + (size_t)nb * C_;
      for (int k0 = 0; k0 < C_; k0 += 32) {
        v16h a0 = load_a16(A, C_, row0,      k0, lane);
        v16h a1 = load_a16(A, C_, row0 + 16, k0, lane);
        v16h f0 = load_row16(rowa, k0, lane, va);
        v16h f1 = load_row16(rowb, k0, lane, vb);
        acc00 = wmma16(a0, f0, acc00);
        acc01 = wmma16(a0, f1, acc01);
        acc10 = wmma16(a1, f0, acc10);
        acc11 = wmma16(a1, f1, acc11);
      }
    }
  }
#pragma unroll
  for (int i = 0; i < 8; ++i) {
    int o0 = row0 + i + ((lane >> 4) << 3);
    int o1 = o0 + 16;
    float inv0 = rsqrtf(var[o0] + 1e-5f);
    float inv1 = rsqrtf(var[o1] + 1e-5f);
    float g0 = inv0 * gamma[o0], g1 = inv1 * gamma[o1];
    float y00 = (acc00[i] - mean[o0]) * g0 + beta[o0];
    float y01 = (acc01[i] - mean[o0]) * g0 + beta[o0];
    float y10 = (acc10[i] - mean[o1]) * g1 + beta[o1];
    float y11 = (acc11[i] - mean[o1]) * g1 + beta[o1];
    size_t base0 = ((size_t)b * OUT_ + o0) * N_;
    size_t base1 = ((size_t)b * OUT_ + o1) * N_;
    out[base0 + n0] = fmaxf(y00, 0.f);
    out[base0 + n1] = fmaxf(y01, 0.f);
    out[base1 + n0] = fmaxf(y10, 0.f);
    out[base1 + n1] = fmaxf(y11, 0.f);
  }
}

// ---------------------------------------------------------------------------
extern "C" void kernel_launch(void* const* d_in, const int* in_sizes, int n_in,
                              void* d_out, int out_size, void* d_ws, size_t ws_size,
                              hipStream_t stream) {
  const float* x     = (const float*)d_in[0];
  const float* wq    = (const float*)d_in[1];
  const float* bq    = (const float*)d_in[2];
  const float* wk    = (const float*)d_in[3];
  const float* bk    = (const float*)d_in[4];
  const float* wv    = (const float*)d_in[5];
  const float* bv    = (const float*)d_in[6];
  const float* w1    = (const float*)d_in[7];
  const float* w2    = (const float*)d_in[8];
  const float* cw    = (const float*)d_in[9];
  const float* gamma = (const float*)d_in[10];
  const float* beta  = (const float*)d_in[11];
  const float* mean  = (const float*)d_in[12];
  const float* var   = (const float*)d_in[13];
  float* out = (float*)d_out;

  char* ws = (char*)d_ws;
  size_t off = 0;
  auto alloc = [&](size_t bytes) -> char* {
    char* p = ws + off;
    off = (off + bytes + 255) & ~(size_t)255;
    return p;
  };
  _Float16* x16  = (_Float16*)alloc((size_t)B_ * C_ * N_ * 2);        // 16.8 MB
  _Float16* xT16 = (_Float16*)alloc((size_t)B_ * N_ * C_ * 2);        // 16.8 MB
  _Float16* q16  = (_Float16*)alloc((size_t)B_ * N_ * C8_ * 2);       //  2.1 MB
  _Float16* kT16 = (_Float16*)alloc((size_t)B_ * N_ * C8_ * 2);       //  2.1 MB
  _Float16* vbuf = (_Float16*)alloc((size_t)B_ * C_ * N_ * 2);        // 16.8 MB
  _Float16* wq16 = (_Float16*)alloc((size_t)C8_ * C_ * 2);
  _Float16* wk16 = (_Float16*)alloc((size_t)C8_ * C_ * 2);
  _Float16* wv16 = (_Float16*)alloc((size_t)C_ * C_ * 2);
  _Float16* wc16 = (_Float16*)alloc((size_t)9 * OUT_ * C_ * 2);
  _Float16* attn = (_Float16*)alloc((size_t)B_ * N_ * N_ * 2);        // 134 MB (energy->attn in place)
  float*    pa   = (float*)   alloc((size_t)B_ * C_ * N_ * 4);        // 33.6 MB
  _Float16* caT  = (_Float16*)alloc((size_t)B_ * N_ * C_ * 2);        // 16.8 MB
  float*    avg  = (float*)   alloc((size_t)B_ * C_ * 4);
  float*    mxp  = (float*)   alloc((size_t)B_ * C_ * 4);
  float*    scl  = (float*)   alloc((size_t)B_ * C_ * 4);
  (void)ws_size; (void)in_sizes; (void)n_in; (void)out_size;

  const size_t total = (size_t)B_ * C_ * N_;            // 8,388,608
  k_convert<<<dim3((unsigned)((total + 255) / 256)), dim3(256), 0, stream>>>(x, x16, xT16);
  k_pack<<<dim3((OUT_ * C_ * 9 + 255) / 256), dim3(256), 0, stream>>>(wq, wk, wv, cw,
                                                                      wq16, wk16, wv16, wc16);
  // q/k projections: M=N_ (4096), Ncols=64
  k_qk<<<dim3(C8_ / 16, N_ / 128, B_), dim3(256), 0, stream>>>(xT16, wq16, wk16, bq, bk, q16, kT16);
  // v projection: M=C (512), Ncols=N_
  k_v<<<dim3(N_ / 16, C_ / 128, B_), dim3(256), 0, stream>>>(xT16, wv16, bv, vbuf);
  // energy: M=N_, Ncols=N_, K=64
  k_energy<<<dim3(N_ / 16, N_ / 128, B_), dim3(256), 0, stream>>>(q16, kT16, attn);
  // softmax in place, one block per row
  k_softmax<<<dim3(B_ * N_), dim3(256), 0, stream>>>(attn);
  // pa: M=C (512), Ncols=N_, K=N_  (+ residual), 32x32 per wave
  k_pa<<<dim3(N_ / 32, C_ / 256, B_), dim3(256), 0, stream>>>(vbuf, attn, x, pa);
  // pooling + SE MLP
  k_pool<<<dim3(B_ * C_), dim3(256), 0, stream>>>(pa, avg, mxp);
  k_mlp<<<dim3(B_), dim3(512), 0, stream>>>(avg, mxp, w1, w2, scl);
  // scaled transpose for conv
  k_caT<<<dim3((unsigned)((total + 255) / 256)), dim3(256), 0, stream>>>(pa, scl, caT);
  // conv 3x3 + BN + ReLU: M=OUT (256, 8 waves*32), Ncols=N_, K=9*512, 32x32 per wave
  k_conv<<<dim3(N_ / 32, 1, B_), dim3(256), 0, stream>>>(wc16, caT, gamma, beta,
                                                         mean, var, out);
}
